// kerl_6313601925615
// MI455X (gfx1250) — matrix operand order repaired and checked by reading, not compiled, not run on hardware
//
#include <hip/hip_runtime.h>

// ---- sizes from the reference ----
#define BB   1024
#define LL   50
#define DD   128
#define KGD  50
#define VV   100000
#define TWO_D 256
#define MLP_IN 228   // D + 2*KG

typedef float  v2f  __attribute__((ext_vector_type(2)));
typedef float  v8f  __attribute__((ext_vector_type(8)));
typedef __bf16 v8bf  __attribute__((ext_vector_type(8)));
typedef __bf16 v16bf __attribute__((ext_vector_type(16)));

// ------------------------------------------------------------------
// 1) BatchNorm stats over kg_map: mu[k], invstd[k] for k in [0,50)
// ------------------------------------------------------------------
__global__ __launch_bounds__(256) void kgstat_kernel(
    const float* __restrict__ kg, float* __restrict__ stats)
{
    __shared__ float s1[256], s2[256];
    const int k = blockIdx.x;            // column 0..49
    float a = 0.f, b = 0.f;
    for (int i = threadIdx.x; i < VV; i += 256) {
        float v = kg[(size_t)i * KGD + k];
        a += v; b += v * v;
    }
    s1[threadIdx.x] = a; s2[threadIdx.x] = b;
    __syncthreads();
    for (int s = 128; s > 0; s >>= 1) {
        if (threadIdx.x < s) {
            s1[threadIdx.x] += s1[threadIdx.x + s];
            s2[threadIdx.x] += s2[threadIdx.x + s];
        }
        __syncthreads();
    }
    if (threadIdx.x == 0) {
        float mu  = s1[0] / (float)VV;
        float var = s2[0] / (float)VV - mu * mu;
        stats[k]      = mu;
        stats[64 + k] = rsqrtf(var + 1e-5f);
    }
}

// ------------------------------------------------------------------
// 2) batch_kg = sum_t kg_bn[seq[b,t]] / train_len[b]; kgh = bkg@mlphW^T+b
// ------------------------------------------------------------------
__global__ __launch_bounds__(64) void bkg_kernel(
    const int* __restrict__ seq, const int* __restrict__ tlen,
    const float* __restrict__ kg, const float* __restrict__ stats,
    const float* __restrict__ mW, const float* __restrict__ mb,
    float* __restrict__ bkg, float* __restrict__ kgh)
{
    __shared__ float s_bkg[KGD];
    const int b = blockIdx.x;
    const int k = threadIdx.x;
    if (k < KGD) {
        const float mu = stats[k], is = stats[64 + k];
        float acc = 0.f;
        for (int t = 0; t < LL; ++t) {
            int idx = seq[b * LL + t];
            acc += (kg[(size_t)idx * KGD + k] - mu) * is;
        }
        acc /= (float)tlen[b];
        bkg[b * KGD + k] = acc;
        s_bkg[k] = acc;
    }
    __syncthreads();
    if (k < KGD) {
        float acc = mb[k];
        for (int q = 0; q < KGD; ++q) acc += s_bkg[q] * mW[k * KGD + q];
        kgh[b * KGD + k] = acc;
    }
}

// ------------------------------------------------------------------
// 3) GRU: 16 batch rows per block, full fp32 via V_WMMA_F32_16X16X4_F32.
//    Wave w owns gate-dim slice [16w, 16w+16) of r/z/n for both GEMMs.
// ------------------------------------------------------------------
#define XS 140   // LDS row stride (floats): 16B-aligned rows, conflict-free
__global__ __launch_bounds__(256) void gru_kernel(
    const int* __restrict__ seq, const int* __restrict__ tlen,
    const float* __restrict__ item_emb,
    const float* __restrict__ Wih, const float* __restrict__ Whh,
    const float* __restrict__ bih, const float* __restrict__ bhh,
    float* __restrict__ H)
{
    __shared__ float x_lds[16 * XS];
    __shared__ float h_lds[16 * XS];
    __shared__ int   len_l[16];

    const int tid    = threadIdx.x;
    const int b0     = blockIdx.x * 16;
    const int wave   = tid >> 5;
    const int lane   = tid & 31;
    const int lane16 = lane & 15;
    const int half   = lane >> 4;
    const int colw   = wave * 16 + lane16;   // hidden dim 0..127
    const int koffB  = half ? 2 : 0;

    for (int i = tid; i < 16 * XS; i += 256) h_lds[i] = 0.f;
    if (tid < 16) len_l[tid] = tlen[b0 + tid];

    const int m_g = tid >> 4;          // gather row 0..15
    const int k_g = (tid & 15) * 8;    // gather col 0..120

    const float bxr = bih[0 * DD + colw], bxz = bih[1 * DD + colw], bxn = bih[2 * DD + colw];
    const float bhr = bhh[0 * DD + colw], bhz = bhh[1 * DD + colw], bhn = bhh[2 * DD + colw];

    const float* wr = Wih + (size_t)(0 * DD + colw) * DD;
    const float* wz = Wih + (size_t)(1 * DD + colw) * DD;
    const float* wn = Wih + (size_t)(2 * DD + colw) * DD;
    const float* ur = Whh + (size_t)(0 * DD + colw) * DD;
    const float* uz = Whh + (size_t)(1 * DD + colw) * DD;
    const float* un = Whh + (size_t)(2 * DD + colw) * DD;

    __syncthreads();

    for (int t = 0; t < LL; ++t) {
        // gather x_t rows into LDS
        {
            int idx = seq[(b0 + m_g) * LL + t];
            const float4* src = (const float4*)(item_emb + (size_t)idx * DD + k_g);
            float4 a = src[0], c = src[1];
            float4* dst = (float4*)(&x_lds[m_g * XS + k_g]);
            dst[0] = a; dst[1] = c;
        }
        __syncthreads();   // x ready; previous-step h writes visible

        v8f axr, axz, axn, ahr, ahz, ahn;
        #pragma unroll
        for (int i = 0; i < 8; ++i) {
            axr[i] = bxr; axz[i] = bxz; axn[i] = bxn;
            ahr[i] = bhr; ahz[i] = bhz; ahn[i] = bhn;
        }

        #pragma unroll 4
        for (int kb = 0; kb < 32; ++kb) {
            const int kk = kb * 4 + koffB;
            v2f ax = *(const v2f*)&x_lds[lane16 * XS + kk];
            v2f ah = *(const v2f*)&h_lds[lane16 * XS + kk];
            v2f br = *(const v2f*)(wr + kk);
            v2f bz = *(const v2f*)(wz + kk);
            v2f bn = *(const v2f*)(wn + kk);
            v2f cr = *(const v2f*)(ur + kk);
            v2f cz = *(const v2f*)(uz + kk);
            v2f cn = *(const v2f*)(un + kk);
            axr = __builtin_amdgcn_wmma_f32_16x16x4_f32(false, ax, false, br, (short)0, axr, false, false);
            axz = __builtin_amdgcn_wmma_f32_16x16x4_f32(false, ax, false, bz, (short)0, axz, false, false);
            axn = __builtin_amdgcn_wmma_f32_16x16x4_f32(false, ax, false, bn, (short)0, axn, false, false);
            ahr = __builtin_amdgcn_wmma_f32_16x16x4_f32(false, ah, false, cr, (short)0, ahr, false, false);
            ahz = __builtin_amdgcn_wmma_f32_16x16x4_f32(false, ah, false, cz, (short)0, ahz, false, false);
            ahn = __builtin_amdgcn_wmma_f32_16x16x4_f32(false, ah, false, cn, (short)0, ahn, false, false);
        }

        float hnew[8];
        #pragma unroll
        for (int r = 0; r < 8; ++r) {
            const int m = r + half * 8;              // D-matrix row mapping
            float rr = 1.f / (1.f + __expf(-(axr[r] + ahr[r])));
            float zz = 1.f / (1.f + __expf(-(axz[r] + ahz[r])));
            float nn = tanhf(axn[r] + rr * ahn[r]);
            float hold = h_lds[m * XS + colw];
            float hv = (1.f - zz) * nn + zz * hold;
            hnew[r] = (t < len_l[m]) ? hv : hold;    // packed-seq freeze
        }
        __syncthreads();   // all waves done reading h_lds
        #pragma unroll
        for (int r = 0; r < 8; ++r)
            h_lds[(r + half * 8) * XS + colw] = hnew[r];
    }
    __syncthreads();
    {
        float4* dst = (float4*)(H + (size_t)(b0 + m_g) * DD + k_g);
        const float4* src = (const float4*)&h_lds[m_g * XS + k_g];
        dst[0] = src[0]; dst[1] = src[1];
    }
}

// ------------------------------------------------------------------
// 4) hidden = tanh([h|bkg|kgh] @ mlp_W^T + mlp_b), stored as bf16
// ------------------------------------------------------------------
__global__ __launch_bounds__(256) void mlp_kernel(
    const float* __restrict__ H, const float* __restrict__ bkg,
    const float* __restrict__ kgh, const float* __restrict__ mlpW,
    const float* __restrict__ mlpb, __bf16* __restrict__ hid)
{
    __shared__ float s_in[MLP_IN];
    const int b = blockIdx.x;
    const int j = threadIdx.x;
    if (j < DD)            s_in[j] = H[b * DD + j];
    else if (j < DD + KGD) s_in[j] = bkg[b * KGD + (j - DD)];
    else if (j < MLP_IN)   s_in[j] = kgh[b * KGD + (j - DD - KGD)];
    __syncthreads();
    const float* w = mlpW + (size_t)j * MLP_IN;
    float acc = mlpb[j];
    #pragma unroll 4
    for (int k = 0; k < MLP_IN; ++k) acc += s_in[k] * w[k];
    hid[b * TWO_D + j] = (__bf16)tanhf(acc);
}

// ------------------------------------------------------------------
// 5) out = hidden @ fc_W^T + fc_b via V_WMMA_F32_16X16X32_BF16.
//    Wave owns 16 output columns; fc_W tile pinned in VGPRs (read once).
// ------------------------------------------------------------------
#define N_WTILES 6250   // V / 16
__global__ __launch_bounds__(256) void fc_kernel(
    const __bf16* __restrict__ hid, const float* __restrict__ fcW,
    const float* __restrict__ fcb, float* __restrict__ out)
{
    const int wave   = threadIdx.x >> 5;
    const int lane   = threadIdx.x & 31;
    const int lane16 = lane & 15;
    const int half   = lane >> 4;
    const int g      = blockIdx.x * 8 + wave;
    if (g >= N_WTILES) return;                 // uniform per-wave exit
    const int col = g * 16 + lane16;           // output column (N)

    // B fragments: fc_W row `col` (K=256) -> 8 x v16bf, same k order as A:
    // lane = N, half-wave supplies K ranges [kb*32+16*half, +16) contiguously.
    v16bf bfrag[8];
    #pragma unroll
    for (int kb = 0; kb < 8; ++kb) {
        const float* wp = fcW + (size_t)col * TWO_D + kb * 32 + half * 16;
        v16bf bb;
        #pragma unroll
        for (int e = 0; e < 16; ++e) bb[e] = (__bf16)wp[e];
        bfrag[kb] = bb;
    }
    const float bias = fcb[col];

    for (int m0 = 0; m0 < BB; m0 += 16) {
        const int arow = m0 + lane16;          // A row (M) = lane
        v8f acc;
        #pragma unroll
        for (int i = 0; i < 8; ++i) acc[i] = bias;
        #pragma unroll
        for (int kb = 0; kb < 8; ++kb) {
            // A layout: lane=M; K slots: [kb*32+16*half, +16) contiguously
            const __bf16* p0 = hid + (size_t)arow * TWO_D + kb * 32 + half * 16;
            v8bf lo = *(const v8bf*)p0;
            v8bf hi = *(const v8bf*)(p0 + 8);
            v16bf a = __builtin_shufflevector(lo, hi,
                0,1,2,3,4,5,6,7,8,9,10,11,12,13,14,15);
            acc = __builtin_amdgcn_wmma_f32_16x16x32_bf16(
                false, a, false, bfrag[kb], (short)0, acc, false, false);
        }
        #pragma unroll
        for (int r = 0; r < 8; ++r) {          // D layout: reg r -> M = r + 8*half
            const int row = m0 + r + half * 8;
            out[(size_t)row * VV + col] = acc[r];
        }
    }
}

// ------------------------------------------------------------------
extern "C" void kernel_launch(void* const* d_in, const int* in_sizes, int n_in,
                              void* d_out, int out_size, void* d_ws, size_t ws_size,
                              hipStream_t stream)
{
    const int*   seq      = (const int*)d_in[0];
    const int*   tlen     = (const int*)d_in[1];
    const float* item_emb = (const float*)d_in[2];
    const float* kg_map   = (const float*)d_in[3];
    const float* gru_Wih  = (const float*)d_in[4];
    const float* gru_Whh  = (const float*)d_in[5];
    const float* gru_bih  = (const float*)d_in[6];
    const float* gru_bhh  = (const float*)d_in[7];
    const float* mlph_W   = (const float*)d_in[8];
    const float* mlph_b   = (const float*)d_in[9];
    const float* mlp_W    = (const float*)d_in[10];
    const float* mlp_b    = (const float*)d_in[11];
    const float* fc_W     = (const float*)d_in[12];
    const float* fc_b     = (const float*)d_in[13];
    float* out = (float*)d_out;

    // workspace layout (all regions fully written before read)
    float*  ws    = (float*)d_ws;
    float*  stats = ws;                       // 128 floats (mu | invstd)
    float*  bkg   = ws + 128;                 // B*KG
    float*  kgh   = bkg + (size_t)BB * KGD;   // B*KG
    float*  H     = kgh + (size_t)BB * KGD;   // B*D
    __bf16* hid   = (__bf16*)(H + (size_t)BB * DD);  // B*2D bf16 (16B aligned)

    kgstat_kernel<<<KGD, 256, 0, stream>>>(kg_map, stats);
    bkg_kernel<<<BB, 64, 0, stream>>>(seq, tlen, kg_map, stats, mlph_W, mlph_b, bkg, kgh);
    gru_kernel<<<BB / 16, 256, 0, stream>>>(seq, tlen, item_emb, gru_Wih, gru_Whh,
                                            gru_bih, gru_bhh, H);
    mlp_kernel<<<BB, 256, 0, stream>>>(H, bkg, kgh, mlp_W, mlp_b, hid);
    fc_kernel<<<(N_WTILES + 7) / 8, 256, 0, stream>>>(hid, fc_W, fc_b, out);
}
